// ABCD_38431367364661
// MI455X (gfx1250) — matrix operand order repaired and checked by reading, not compiled, not run on hardware
//
#include <hip/hip_runtime.h>
#include <hip/hip_bf16.h>
#include <math.h>

typedef __attribute__((ext_vector_type(16))) _Float16 v16h;
typedef __attribute__((ext_vector_type(8)))  float    v8f;

#define BB    8
#define HD    128
#define WD    128
#define CE    64
#define HIDN  256
#define QN    16384
#define MTOT  (BB * QN)      /* 131072 */
#define KPAD0 288            /* 257 padded to multiple of 32 */

// ---------------------------------------------------------------------------
// WMMA fragment loaders matching the CDNA5 ISA VGPR layouts (wave32).
// A (16x32 f16, row-major [M][K]): lane m=lane&15, kbase=8*(lane>>4);
//   halves 0..7 = A[m][kbase+0..7], halves 8..15 = A[m][kbase+16..23]
// B (32x16 f16, stored transposed as [N][K]): lane n=lane&15, kbase=8*(lane>>4);
//   halves 0..7 = Bt[n][kbase+0..7], halves 8..15 = Bt[n][kbase+16..23]
// C/D (16x16 f32): VGPR r -> M = r + 8*(lane>>4), N = lane&15
// ---------------------------------------------------------------------------
__device__ __forceinline__ v16h load_a16(const _Float16* __restrict__ p) {
  v16h a;
#pragma unroll
  for (int i = 0; i < 8; ++i) a[i] = p[i];
#pragma unroll
  for (int i = 0; i < 8; ++i) a[8 + i] = p[16 + i];
  return a;
}
__device__ __forceinline__ v16h load_b16(const _Float16* __restrict__ p) {
  v16h b;
#pragma unroll
  for (int i = 0; i < 8; ++i) b[i] = p[i];
#pragma unroll
  for (int i = 0; i < 8; ++i) b[8 + i] = p[16 + i];
  return b;
}
#define WMMA32(acc, a, b) \
  __builtin_amdgcn_wmma_f32_16x16x32_f16(false, (a), false, (b), (short)0, (acc), false, false)

// ---------------------------------------------------------------------------
// Weight reorder kernels (f32 -> f16, per launch; deterministic)
// ---------------------------------------------------------------------------
// conv weights [O][I][3][3] f32 -> [O][(ty*3+tx)*I + i] f16 (K matches NHWC patches)
__global__ void k_conv_w_cvt(const float* __restrict__ src, _Float16* __restrict__ dst,
                             int O, int I) {
  int idx = blockIdx.x * blockDim.x + threadIdx.x;
  int total = O * I * 9;
  if (idx >= total) return;
  int tx = idx % 3; int t = idx / 3;
  int ty = t % 3;   t /= 3;
  int i  = t % I;   int o = t / I;
  dst[(size_t)o * (9 * I) + (ty * 3 + tx) * I + i] = (_Float16)src[idx];
}

// mlp weights [K][N] f32 -> [N][Kpad] f16 (transpose + zero pad)
__global__ void k_mlp_w_cvt(const float* __restrict__ src, _Float16* __restrict__ dst,
                            int K, int N, int Kpad) {
  int idx = blockIdx.x * blockDim.x + threadIdx.x;
  if (idx >= N * Kpad) return;
  int k = idx % Kpad; int n = idx / Kpad;
  dst[idx] = (_Float16)((k < K) ? src[(size_t)k * N + n] : 0.0f);
}

// ---------------------------------------------------------------------------
// Head conv 3->64 (K=27, too small for WMMA): scalar, NCHW f32 in -> NHWC f16 out
// ---------------------------------------------------------------------------
__global__ void k_head_conv(const float* __restrict__ inp, const float* __restrict__ w,
                            const float* __restrict__ bias, _Float16* __restrict__ Y) {
  int idx = blockIdx.x * blockDim.x + threadIdx.x;
  if (idx >= BB * HD * WD * CE) return;
  int o = idx & 63;
  int p = idx >> 6;
  int x = p % WD; int t = p / WD;
  int h = t % HD; int b = t / HD;
  float acc = bias[o];
#pragma unroll
  for (int i = 0; i < 3; ++i)
#pragma unroll
    for (int ty = 0; ty < 3; ++ty) {
      int hh = h + ty - 1;
      if (hh < 0 || hh >= HD) continue;
#pragma unroll
      for (int tx = 0; tx < 3; ++tx) {
        int ww = x + tx - 1;
        if (ww < 0 || ww >= WD) continue;
        acc += inp[(((size_t)b * 3 + i) * HD + hh) * WD + ww] *
               w[((o * 3 + i) * 3 + ty) * 3 + tx];
      }
    }
  Y[idx] = (_Float16)acc;
}

// ---------------------------------------------------------------------------
// 3x3 conv, CIN=64, implicit GEMM via WMMA.  One wave computes a 16-pixel x
// 64-outchannel tile (4 accumulators sharing each A fragment).  K = 9 taps *
// 64 ch = 18 A fragments, 72 WMMAs per wave.  Wave-uniform interior fast path
// (no bounds checks) vs. guarded border path; EXEC is full at every WMMA.
// X: NHWC f16 [B][H][W][64]; Wt: [OC][576] f16; Y: NHWC f16 [B][H][W][OC].
// ---------------------------------------------------------------------------
__global__ void k_conv3x3_wmma(const _Float16* __restrict__ X,
                               const _Float16* __restrict__ Wt,
                               const float* __restrict__ bias,
                               const _Float16* __restrict__ skip,
                               _Float16* __restrict__ Y, int OC, int relu) {
  const int lane = threadIdx.x & 31;
  const int lo = lane & 15;
  const int hi = lane >> 4;
  const int mt = blockIdx.x;                  // pixel tile (16 pixels in a row)
  const int ng = blockIdx.y;                  // group of 4 out-channel tiles
  const int b  = mt / ((HD * WD) / 16);
  const int pt = mt % ((HD * WD) / 16);
  const int h  = pt / (WD / 16);
  const int w0 = (pt % (WD / 16)) * 16;

  const _Float16* brow[4];
#pragma unroll
  for (int j = 0; j < 4; ++j)
    brow[j] = Wt + (size_t)((ng * 4 + j) * 16 + lo) * (9 * CE) + 8 * hi;

  v8f acc[4] = {{}, {}, {}, {}};

  const bool interior = (h >= 1) && (h <= HD - 2) && (w0 >= 16) && (w0 <= WD - 32);
  if (interior) {
    // wave-uniform fast path: all 3x3 taps in-bounds for every lane
    const _Float16* base = X + (((size_t)b * HD + h) * WD + w0 + lo) * CE + 8 * hi;
#pragma unroll
    for (int ty = 0; ty < 3; ++ty)
#pragma unroll
      for (int tx = 0; tx < 3; ++tx) {
        const _Float16* arow = base + ((ty - 1) * WD + (tx - 1)) * CE;
        const int kb = (ty * 3 + tx) * CE;
#pragma unroll
        for (int kt = 0; kt < CE / 32; ++kt) {
          v16h a = load_a16(arow + kt * 32);
#pragma unroll
          for (int j = 0; j < 4; ++j)
            acc[j] = WMMA32(acc[j], a, load_b16(brow[j] + kb + kt * 32));
        }
      }
  } else {
    const int w = w0 + lo;
#pragma unroll
    for (int ty = 0; ty < 3; ++ty) {
      int hh = h + ty - 1;
      bool hok = (hh >= 0) && (hh < HD);
#pragma unroll
      for (int tx = 0; tx < 3; ++tx) {
        int ww = w + tx - 1;
        bool ok = hok && (ww >= 0) && (ww < WD);
        const _Float16* arow =
            X + (((size_t)b * HD + (hok ? hh : 0)) * WD + (ok ? ww : 0)) * CE + 8 * hi;
        const int kb = (ty * 3 + tx) * CE;
#pragma unroll
        for (int kt = 0; kt < CE / 32; ++kt) {
          v16h a = {};
          if (ok) a = load_a16(arow + kt * 32);
#pragma unroll
          for (int j = 0; j < 4; ++j)
            acc[j] = WMMA32(acc[j], a, load_b16(brow[j] + kb + kt * 32));
        }
      }
    }
  }

  // epilogue: D layout -> pixel = w0 + r + 8*hi, channel = (ng*4+j)*16 + lo
#pragma unroll
  for (int j = 0; j < 4; ++j) {
    const int n = (ng * 4 + j) * 16 + lo;
    const float bv = bias[n];
#pragma unroll
    for (int r = 0; r < 8; ++r) {
      int px = w0 + r + 8 * hi;
      size_t off = (((size_t)b * HD + h) * WD + px) * OC + n;
      float v = acc[j][r] + bv;
      if (skip) v += (float)skip[off];
      if (relu) v = fmaxf(v, 0.0f);
      Y[off] = (_Float16)v;
    }
  }
}

// ---------------------------------------------------------------------------
// Gather (grid_sample nearest, align_corners=False, zero pad) + Fourier
// modulation + cell concat.  Builds H0 [MTOT][KPAD0] f16.
// feature[c] = coef[c] * (c<128 ? cos(pi*freq[2c]) : sin(pi*freq[2(c-128)+1]))
// ---------------------------------------------------------------------------
__global__ void k_gather_fourier(const _Float16* __restrict__ coef,
                                 const _Float16* __restrict__ freq,
                                 const float* __restrict__ coord,
                                 const float* __restrict__ cell,
                                 _Float16* __restrict__ H0) {
  const int m = blockIdx.x;        // query index 0..MTOT-1
  const int c = threadIdx.x;       // 0..KPAD0-1
  const int b = m >> 14;
  const float cy = coord[(size_t)m * 2 + 0];
  const float cx = coord[(size_t)m * 2 + 1];
  const int iy = (int)rintf(((cy + 1.0f) * HD - 1.0f) * 0.5f);
  const int ix = (int)rintf(((cx + 1.0f) * WD - 1.0f) * 0.5f);
  const bool valid = (iy >= 0) && (iy < HD) && (ix >= 0) && (ix < WD);
  float v = 0.0f;
  if (c < HIDN) {
    if (valid) {
      size_t base = (((size_t)b * HD + iy) * WD + ix) * HIDN;
      float cf = (float)coef[base + c];
      float fr;
      if (c < HIDN / 2)
        fr = cosf(3.14159265358979f * (float)freq[base + 2 * c]);
      else
        fr = sinf(3.14159265358979f * (float)freq[base + 2 * (c - HIDN / 2) + 1]);
      v = cf * fr;
    }
  } else if (c == HIDN) {
    v = cell[m];
  }
  H0[(size_t)m * KPAD0 + c] = (_Float16)v;
}

// ---------------------------------------------------------------------------
// MLP GEMM: C[M][256] = relu(A[M][lda] * Wt[N][ldb]^T + bias).  One wave per
// 16x64 tile: 4 accumulators share each A fragment; k-loop fully unrolled via
// the KT template parameter (KT*4 straight-line WMMAs per wave).
// ---------------------------------------------------------------------------
template <int KT>
__global__ void k_gemm_wmma(const _Float16* __restrict__ A, int lda,
                            const _Float16* __restrict__ Wt, int ldb,
                            const float* __restrict__ bias,
                            _Float16* __restrict__ C, int ldc, int relu) {
  const int lane = threadIdx.x & 31;
  const int lo = lane & 15;
  const int hi = lane >> 4;
  const int mt = blockIdx.x;
  const int ng = blockIdx.y;                  // group of 4 n-tiles
  const _Float16* arow = A + (size_t)(mt * 16 + lo) * lda + 8 * hi;
  const _Float16* brow[4];
#pragma unroll
  for (int j = 0; j < 4; ++j)
    brow[j] = Wt + (size_t)((ng * 4 + j) * 16 + lo) * ldb + 8 * hi;

  v8f acc[4] = {{}, {}, {}, {}};
#pragma unroll
  for (int k = 0; k < KT; ++k) {
    v16h a = load_a16(arow + k * 32);
#pragma unroll
    for (int j = 0; j < 4; ++j)
      acc[j] = WMMA32(acc[j], a, load_b16(brow[j] + k * 32));
  }

  const int mbase = mt * 16 + 8 * hi;
#pragma unroll
  for (int j = 0; j < 4; ++j) {
    const int n = (ng * 4 + j) * 16 + lo;
    const float bv = bias[n];
#pragma unroll
    for (int r = 0; r < 8; ++r) {
      float v = acc[j][r] + bv;
      if (relu) v = fmaxf(v, 0.0f);
      C[(size_t)(mbase + r) * ldc + n] = (_Float16)v;
    }
  }
}

// ---------------------------------------------------------------------------
// Final 256->3 layer + tanh scaling, scalar (N=3 too small for WMMA).
// ---------------------------------------------------------------------------
__global__ void k_final(const _Float16* __restrict__ H, const float* __restrict__ w4,
                        const float* __restrict__ b4, float* __restrict__ out) {
  int idx = blockIdx.x * blockDim.x + threadIdx.x;
  if (idx >= MTOT * 3) return;
  int j = idx % 3; int m = idx / 3;
  const _Float16* h = H + (size_t)m * HIDN;
  float acc = b4[j];
#pragma unroll 8
  for (int k = 0; k < HIDN; ++k) acc += (float)h[k] * w4[k * 3 + j];
  out[idx] = (tanhf(acc) + 1.0f) * 0.5f;
}

// ---------------------------------------------------------------------------
extern "C" void kernel_launch(void* const* d_in, const int* in_sizes, int n_in,
                              void* d_out, int out_size, void* d_ws, size_t ws_size,
                              hipStream_t stream) {
  (void)in_sizes; (void)n_in; (void)out_size; (void)ws_size;
  const float* inp    = (const float*)d_in[0];
  const float* coord  = (const float*)d_in[1];
  const float* cell   = (const float*)d_in[2];
  const float* head_w = (const float*)d_in[3];
  const float* head_b = (const float*)d_in[4];
  const float* rbw[4] = {(const float*)d_in[5], (const float*)d_in[7],
                         (const float*)d_in[9], (const float*)d_in[11]};
  const float* rbb[4] = {(const float*)d_in[6], (const float*)d_in[8],
                         (const float*)d_in[10], (const float*)d_in[12]};
  const float* coef_w = (const float*)d_in[13];
  const float* coef_b = (const float*)d_in[14];
  const float* freq_w = (const float*)d_in[15];
  const float* freq_b = (const float*)d_in[16];
  const float* mw[4]  = {(const float*)d_in[17], (const float*)d_in[19],
                         (const float*)d_in[21], (const float*)d_in[23]};
  const float* mb[4]  = {(const float*)d_in[18], (const float*)d_in[20],
                         (const float*)d_in[22], (const float*)d_in[24]};
  const float* mlp_w4 = (const float*)d_in[25];
  const float* mlp_b4 = (const float*)d_in[26];
  float* out = (float*)d_out;

  // ---- workspace layout (f16 elements) ----
  const size_t SX  = (size_t)BB * HD * WD * CE;    // encoder feature
  const size_t SC  = (size_t)BB * HD * WD * HIDN;  // coef/freq (== MTOT*256)
  const size_t SH0 = (size_t)MTOT * KPAD0;         // MLP input
  _Float16* ws   = (_Float16*)d_ws;
  _Float16* bufA = ws;                 // x0, later x2
  _Float16* bufB = ws + SX;            // relu temp
  _Float16* bufC = ws + 2 * SX;        // x1
  _Float16* coefB = ws + 3 * SX;       // coef, later H1/H3
  _Float16* freqB = coefB + SC;        // freq, later H2/H4
  _Float16* H0    = freqB + SC;
  _Float16* wptr  = H0 + SH0;
  _Float16* wrb[4]; for (int i = 0; i < 4; ++i) { wrb[i] = wptr; wptr += (size_t)CE * 576; }
  _Float16* wcoef = wptr; wptr += (size_t)HIDN * 576;
  _Float16* wfreq = wptr; wptr += (size_t)HIDN * 576;
  _Float16* wm0   = wptr; wptr += (size_t)HIDN * KPAD0;
  _Float16* wm[3]; for (int i = 0; i < 3; ++i) { wm[i] = wptr; wptr += (size_t)HIDN * HIDN; }

  // ---- weight conversions ----
  for (int i = 0; i < 4; ++i)
    k_conv_w_cvt<<<(CE * CE * 9 + 255) / 256, 256, 0, stream>>>(rbw[i], wrb[i], CE, CE);
  k_conv_w_cvt<<<(HIDN * CE * 9 + 255) / 256, 256, 0, stream>>>(coef_w, wcoef, HIDN, CE);
  k_conv_w_cvt<<<(HIDN * CE * 9 + 255) / 256, 256, 0, stream>>>(freq_w, wfreq, HIDN, CE);
  k_mlp_w_cvt<<<(HIDN * KPAD0 + 255) / 256, 256, 0, stream>>>(mw[0], wm0, 257, HIDN, KPAD0);
  for (int i = 0; i < 3; ++i)
    k_mlp_w_cvt<<<(HIDN * HIDN + 255) / 256, 256, 0, stream>>>(mw[i + 1], wm[i], HIDN, HIDN, HIDN);

  // ---- encoder ----
  k_head_conv<<<(BB * HD * WD * CE + 255) / 256, 256, 0, stream>>>(inp, head_w, head_b, bufA);
  const dim3 cgrid(BB * HD * WD / 16, CE / 64);       // 64 out channels: 1 group
  // rb1
  k_conv3x3_wmma<<<cgrid, 32, 0, stream>>>(bufA, wrb[0], rbb[0], nullptr, bufB, CE, 1);
  k_conv3x3_wmma<<<cgrid, 32, 0, stream>>>(bufB, wrb[1], rbb[1], bufA,    bufC, CE, 0);
  // rb2
  k_conv3x3_wmma<<<cgrid, 32, 0, stream>>>(bufC, wrb[2], rbb[2], nullptr, bufB, CE, 1);
  k_conv3x3_wmma<<<cgrid, 32, 0, stream>>>(bufB, wrb[3], rbb[3], bufC,    bufA, CE, 0);
  // coef / freq convs (64 -> 256): 4 groups of 4 n-tiles
  const dim3 cfgrid(BB * HD * WD / 16, HIDN / 64);
  k_conv3x3_wmma<<<cfgrid, 32, 0, stream>>>(bufA, wcoef, coef_b, nullptr, coefB, HIDN, 0);
  k_conv3x3_wmma<<<cfgrid, 32, 0, stream>>>(bufA, wfreq, freq_b, nullptr, freqB, HIDN, 0);

  // ---- gather + fourier -> H0 [MTOT][288] ----
  k_gather_fourier<<<MTOT, KPAD0, 0, stream>>>(coefB, freqB, coord, cell, H0);

  // ---- MLP (coef/freq buffers reused as ping-pong) ----
  const dim3 ggrid(MTOT / 16, HIDN / 64);
  k_gemm_wmma<KPAD0 / 32><<<ggrid, 32, 0, stream>>>(H0,    KPAD0, wm0,   KPAD0, mb[0], coefB, HIDN, 1);
  k_gemm_wmma<HIDN / 32><<<ggrid, 32, 0, stream>>>(coefB, HIDN,  wm[0], HIDN,  mb[1], freqB, HIDN, 1);
  k_gemm_wmma<HIDN / 32><<<ggrid, 32, 0, stream>>>(freqB, HIDN,  wm[1], HIDN,  mb[2], coefB, HIDN, 1);
  k_gemm_wmma<HIDN / 32><<<ggrid, 32, 0, stream>>>(coefB, HIDN,  wm[2], HIDN,  mb[3], freqB, HIDN, 1);

  // ---- final 256->3 + tanh ----
  k_final<<<(MTOT * 3 + 255) / 256, 256, 0, stream>>>(freqB, mlp_w4, mlp_b4, out);
}
